// Corres_Layer_28982439313540
// MI455X (gfx1250) — matrix operand order repaired and checked by reading, not compiled, not run on hardware
//
#include <hip/hip_runtime.h>

typedef __attribute__((ext_vector_type(16))) _Float16 v16h;
typedef __attribute__((ext_vector_type(8)))  _Float16 v8h;
typedef __attribute__((ext_vector_type(8)))  float    v8f;

#define BB 4
#define CC 128
#define NN 4096
#define EPSF 2.220446049250313e-16f
// exp(x/T) = exp2(x * (1/T) * log2(e)), T = 0.01
#define KSCALE 144.26950408889634f

union U16 { v16h v; v8h h[2]; _Float16 e[16]; };

#if __has_builtin(__builtin_amdgcn_global_load_async_to_lds_b128)
#define HAVE_ASYNC_LDS 1
typedef __attribute__((ext_vector_type(4))) int i4;
typedef __attribute__((address_space(1))) i4* gp128;   // global <4 x i32>*
typedef __attribute__((address_space(3))) i4* lp128;   // LDS    <4 x i32>*
#endif

template <int N>
static __device__ __forceinline__ void wait_async() {
#if __has_builtin(__builtin_amdgcn_s_wait_asynccnt)
  __builtin_amdgcn_s_wait_asynccnt(N);
#else
  asm volatile("s_wait_asynccnt %0" :: "i"(N) : "memory");
#endif
}

// ---------------------------------------------------------------------------
// Cross-lane reductions over the 16 lanes of each wave32 half (halves match
// the WMMA C/D row split). v_permlane16_b32 = pure VALU, no LDS round-trips.
// ---------------------------------------------------------------------------
#if __has_builtin(__builtin_amdgcn_permlane16)
static __device__ __forceinline__ float permxor(float x, unsigned s0, unsigned s1) {
  unsigned u = __builtin_bit_cast(unsigned, x);
  u = __builtin_amdgcn_permlane16(u, u, s0, s1, false, false);
  return __builtin_bit_cast(float, u);
}
static __device__ __forceinline__ float red_max16(float x) {
  x = fmaxf(x, permxor(x, 0x67452301u, 0xEFCDAB89u));  // xor 1
  x = fmaxf(x, permxor(x, 0x54761032u, 0xDCFE98BAu));  // xor 2
  x = fmaxf(x, permxor(x, 0x32107654u, 0xBA98FEDCu));  // xor 4
  x = fmaxf(x, permxor(x, 0xFEDCBA98u, 0x76543210u));  // xor 8
  return x;
}
static __device__ __forceinline__ float red_sum16(float x) {
  x += permxor(x, 0x67452301u, 0xEFCDAB89u);
  x += permxor(x, 0x54761032u, 0xDCFE98BAu);
  x += permxor(x, 0x32107654u, 0xBA98FEDCu);
  x += permxor(x, 0xFEDCBA98u, 0x76543210u);
  return x;
}
#else
static __device__ __forceinline__ float red_max16(float x) {
  x = fmaxf(x, __shfl_xor(x, 1, 32));
  x = fmaxf(x, __shfl_xor(x, 2, 32));
  x = fmaxf(x, __shfl_xor(x, 4, 32));
  x = fmaxf(x, __shfl_xor(x, 8, 32));
  return x;
}
static __device__ __forceinline__ float red_sum16(float x) {
  x += __shfl_xor(x, 1, 32);
  x += __shfl_xor(x, 2, 32);
  x += __shfl_xor(x, 4, 32);
  x += __shfl_xor(x, 8, 32);
  return x;
}
#endif

__global__ void zero_means_kernel(float* means) {
  int i = blockIdx.x * 256 + threadIdx.x;
  if (i < 2 * BB * CC) means[i] = 0.0f;
}

// p = W @ x + b (1x1 conv as channel matmul) via f16 WMMA.
// Two o-tiles interleaved per step so the two accumulator chains hide each
// other's WMMA D->C RAW hazards. Atomic spatial sums feed the mean.
__global__ __launch_bounds__(256)
void proj_kernel(const float* __restrict__ m, const float* __restrict__ rs,
                 const float* __restrict__ theta_w, const float* __restrict__ theta_b,
                 const float* __restrict__ phi_w,   const float* __restrict__ phi_b,
                 float* __restrict__ pbuf, float* __restrict__ means) {
  const int wave = threadIdx.x >> 5;
  const int lane = threadIdx.x & 31;
  const int l = lane & 15, hi = lane >> 4;
  const int b = blockIdx.y;
  const int t = blockIdx.z;                 // 0: theta(rs), 1: phi(m)
  const int n0 = blockIdx.x * 128 + wave * 16;

  const float* x    = (t == 0) ? rs : m;
  const float* w    = (t == 0) ? theta_w : phi_w;
  const float* bias = (t == 0) ? theta_b : phi_b;
  float* pout = pbuf + (size_t)t * BB * CC * NN + (size_t)b * CC * NN;
  float* msum = means + t * BB * CC + b * CC;
  const float* xb = x + (size_t)b * CC * NN;

  // B fragments (x): 32x16, lane = col n0+l, per-lane K run of 16 (lo:K0-15, hi:K16-31)
  const int n = n0 + l;
  U16 bf[4];
#pragma unroll
  for (int kf = 0; kf < 4; ++kf) {
    const int c0 = kf * 32 + hi * 16;
#pragma unroll
    for (int kk = 0; kk < 16; ++kk)
      bf[kf].e[kk] = (_Float16)xb[(size_t)(c0 + kk) * NN + n];
  }

  for (int ot = 0; ot < 8; ot += 2) {
    const int o0 = ot * 16, o1 = o0 + 16;
    U16 af0[4], af1[4];
#pragma unroll
    for (int kf = 0; kf < 4; ++kf) {
      const float* wr0 = w + (size_t)(o0 + l) * CC + kf * 32;
      const float* wr1 = w + (size_t)(o1 + l) * CC + kf * 32;
#pragma unroll
      for (int j = 0; j < 8; ++j) {
        af0[kf].e[j]     = (_Float16)wr0[hi * 16 + j];
        af0[kf].e[8 + j] = (_Float16)wr0[hi * 16 + 8 + j];
        af1[kf].e[j]     = (_Float16)wr1[hi * 16 + j];
        af1[kf].e[8 + j] = (_Float16)wr1[hi * 16 + 8 + j];
      }
    }
    v8f sa = {}, sb = {};
#pragma unroll
    for (int kf = 0; kf < 4; ++kf) {
      sa = __builtin_amdgcn_wmma_f32_16x16x32_f16(false, af0[kf].v, false, bf[kf].v,
                                                  (short)0, sa, false, false);
      sb = __builtin_amdgcn_wmma_f32_16x16x32_f16(false, af1[kf].v, false, bf[kf].v,
                                                  (short)0, sb, false, false);
    }
#pragma unroll
    for (int r = 0; r < 8; ++r) {
      const int oa = o0 + r + hi * 8, ob = o1 + r + hi * 8;
      const float va = sa[r] + bias[oa];
      const float vb = sb[r] + bias[ob];
      pout[(size_t)oa * NN + n] = va;
      pout[(size_t)ob * NN + n] = vb;
      const float ra = red_sum16(va);
      const float rb = red_sum16(vb);
      if (l == 0) { atomicAdd(&msum[oa], ra); atomicAdd(&msum[ob], rb); }
    }
  }
}

// Mean-center over spatial, L2-normalize over channels.
// theta/phi written f16 [B][N][C]; rs written f16 [B][C][N].
__global__ __launch_bounds__(256)
void norm_kernel(const float* __restrict__ rs,
                 const float* __restrict__ pbuf, const float* __restrict__ means,
                 _Float16* __restrict__ theta_h, _Float16* __restrict__ phi_h,
                 _Float16* __restrict__ rs_h) {
  const int t = blockIdx.z, b = blockIdx.y;
  const int nn = blockIdx.x * 256 + threadIdx.x;
  __shared__ float sm[CC];
  if (threadIdx.x < CC)
    sm[threadIdx.x] = means[t * BB * CC + b * CC + threadIdx.x] * (1.0f / NN);
  __syncthreads();

  const float* p = pbuf + (size_t)t * BB * CC * NN + (size_t)b * CC * NN;
  float ss = 0.0f;
#pragma unroll 4
  for (int c = 0; c < CC; ++c) {
    const float v = p[(size_t)c * NN + nn] - sm[c];
    ss += v * v;
  }
  const float inv = 1.0f / (sqrtf(ss) + EPSF);
  _Float16* orow = ((t == 0) ? theta_h : phi_h) + ((size_t)b * NN + nn) * CC;
#pragma unroll 4
  for (int c = 0; c < CC; ++c)
    orow[c] = (_Float16)((p[(size_t)c * NN + nn] - sm[c]) * inv);
  if (t == 0) {
    const float* rb = rs + (size_t)b * CC * NN;
    _Float16* rh = rs_h + (size_t)b * CC * NN;
#pragma unroll 4
    for (int c = 0; c < CC; ++c)
      rh[(size_t)c * NN + nn] = (_Float16)rb[(size_t)c * NN + nn];
  }
}

// Cooperative async staging of one 32-key tile: theta (32 x 128 halfs) and
// rs (128 x 32 halfs) -> LDS, 2x b128 each per thread, tracked by ASYNCcnt.
static __device__ __forceinline__ void stage_tiles(const _Float16* __restrict__ thB,
                                                   const _Float16* __restrict__ rsB,
                                                   _Float16* kb, _Float16* vb,
                                                   int j, int tid) {
  const int krow = tid >> 3, kwi = (tid & 7) * 16;          // theta: 32B per thread
  const _Float16* gk = thB + (size_t)(j + krow) * CC + kwi;
  _Float16* sk = kb + krow * CC + kwi;
  const int c = tid >> 1, vwi = (tid & 1) * 16;             // rs: 32B per thread
  const _Float16* gv = rsB + (size_t)c * NN + j + vwi;
  _Float16* sv = vb + c * 32 + vwi;
#if defined(HAVE_ASYNC_LDS)
  // imm offset is added to both global and LDS addresses (ISA 08_async_tensor)
  __builtin_amdgcn_global_load_async_to_lds_b128((gp128)(void*)gk, (lp128)(void*)sk, 0, 0);
  __builtin_amdgcn_global_load_async_to_lds_b128((gp128)(void*)gk, (lp128)(void*)sk, 16, 0);
  __builtin_amdgcn_global_load_async_to_lds_b128((gp128)(void*)gv, (lp128)(void*)sv, 0, 0);
  __builtin_amdgcn_global_load_async_to_lds_b128((gp128)(void*)gv, (lp128)(void*)sv, 16, 0);
#else
  *(v8h*)(sk)     = *(const v8h*)(gk);
  *(v8h*)(sk + 8) = *(const v8h*)(gk + 8);
  *(v8h*)(sv)     = *(const v8h*)(gv);
  *(v8h*)(sv + 8) = *(const v8h*)(gv + 8);
#endif
}

// Fused flash attention with double-buffered async K/V staging in LDS:
// S = phi^T theta (WMMA), online softmax (exp2), denominator via ones-WMMA,
// Y += P @ rs (WMMA). confidence = running max, location = 1/denominator.
__global__ __launch_bounds__(256)
void attn_kernel(const _Float16* __restrict__ theta_h, const _Float16* __restrict__ phi_h,
                 const _Float16* __restrict__ rs_h, float* __restrict__ out) {
  const int tid  = threadIdx.x;
  const int wave = tid >> 5;
  const int lane = tid & 31;
  const int l = lane & 15, hi = lane >> 4;
  const int b = blockIdx.y;
  const int qbase = blockIdx.x * 128 + wave * 16;

  __shared__ __align__(32) _Float16 kbuf[2][32 * CC];   // theta tile [key][c], 2 x 8KB
  __shared__ __align__(32) _Float16 vbuf[2][CC * 32];   // rs tile [c][key], 2 x 8KB
  __shared__ __align__(32) _Float16 pshm[8][16 * 40];   // per-wave 16x32 P tile (+pad)
  _Float16* myp = pshm[wave];

  const _Float16* phiB = phi_h   + (size_t)b * NN * CC;
  const _Float16* thB  = theta_h + (size_t)b * NN * CC;
  const _Float16* rsB  = rs_h    + (size_t)b * CC * NN;

  // Q (A fragments, resident for the whole loop): row = qbase + (lane&15)
  U16 aq[4];
  {
    const _Float16* qr = phiB + (size_t)(qbase + l) * CC;
#pragma unroll
    for (int kf = 0; kf < 4; ++kf) {
      aq[kf].h[0] = *(const v8h*)(qr + kf * 32 + hi * 16);
      aq[kf].h[1] = *(const v8h*)(qr + kf * 32 + hi * 16 + 8);
    }
  }

  // All-ones B fragment: row-sum of P via one WMMA (softmax denominator)
  U16 onesf;
#pragma unroll
  for (int k = 0; k < 16; ++k) onesf.e[k] = (_Float16)1.0f;

  v8f acc[8];
  const v8f vz = {};
#pragma unroll
  for (int cc = 0; cc < 8; ++cc) acc[cc] = vz;
  float mrow[8], lrow[8];
#pragma unroll
  for (int r = 0; r < 8; ++r) { mrow[r] = -3.0e38f; lrow[r] = 0.0f; }

  stage_tiles(thB, rsB, kbuf[0], vbuf[0], 0, tid);

  for (int j = 0; j < NN; j += 32) {
    const int cur = (j >> 5) & 1;
    const bool more = (j + 32 < NN);
    if (more) stage_tiles(thB, rsB, kbuf[cur ^ 1], vbuf[cur ^ 1], j + 32, tid);
    if (more) wait_async<4>(); else wait_async<0>();   // cur tile landed; next in flight
    __syncthreads();

    const _Float16* kb = kbuf[cur];
    const _Float16* vb = vbuf[cur];

    // S tiles for keys [j, j+16) and [j+16, j+32): interleaved chains
    v8f s0 = {}, s1 = {};
#pragma unroll
    for (int kf = 0; kf < 4; ++kf) {
      const v16h bt0 = *(const v16h*)(kb + (size_t)l * CC + kf * 32 + hi * 16);
      const v16h bt1 = *(const v16h*)(kb + (size_t)(16 + l) * CC + kf * 32 + hi * 16);
      s0 = __builtin_amdgcn_wmma_f32_16x16x32_f16(false, aq[kf].v, false, bt0,
                                                  (short)0, s0, false, false);
      s1 = __builtin_amdgcn_wmma_f32_16x16x32_f16(false, aq[kf].v, false, bt1,
                                                  (short)0, s1, false, false);
    }
    // Online softmax per row (row = r + 8*hi; same mapping for all C/D tiles)
    float corr[8];
#pragma unroll
    for (int r = 0; r < 8; ++r) {
      const float mt = red_max16(fmaxf(s0[r], s1[r]));
      const float mn = fmaxf(mrow[r], mt);
      corr[r] = __builtin_exp2f((mrow[r] - mn) * KSCALE);
      mrow[r] = mn;
      const float p0 = __builtin_exp2f((s0[r] - mn) * KSCALE);
      const float p1 = __builtin_exp2f((s1[r] - mn) * KSCALE);
      myp[(r + hi * 8) * 40 + l]      = (_Float16)p0;
      myp[(r + hi * 8) * 40 + 16 + l] = (_Float16)p1;
    }
#pragma unroll
    for (int cc = 0; cc < 8; ++cc)
#pragma unroll
      for (int r = 0; r < 8; ++r) acc[cc][r] *= corr[r];

    // Re-shape P (C/D layout) -> A fragment through the per-wave LDS tile.
    asm volatile("s_wait_dscnt 0" ::: "memory");
    U16 pa;
    pa.h[0] = *(const v8h*)(myp + l * 40 + hi * 16);
    pa.h[1] = *(const v8h*)(myp + l * 40 + hi * 16 + 8);

    // Softmax denominator: row sums of P via one WMMA against ones.
    v8f ssum = {};
    ssum = __builtin_amdgcn_wmma_f32_16x16x32_f16(false, pa.v, false, onesf.v,
                                                  (short)0, ssum, false, false);
#pragma unroll
    for (int r = 0; r < 8; ++r) lrow[r] = lrow[r] * corr[r] + ssum[r];

    // Y(16x128) += P(16x32) @ V(32x128): 8 independent output column chunks
#pragma unroll
    for (int cc = 0; cc < 8; ++cc) {
      const v16h vfrag = *(const v16h*)(vb + (size_t)(cc * 16 + l) * 32 + hi * 16);
      acc[cc] = __builtin_amdgcn_wmma_f32_16x16x32_f16(false, pa.v, false, vfrag,
                                                       (short)0, acc[cc], false, false);
    }
    __syncthreads();   // all waves done with 'cur' before it is restaged
  }

  // Epilogue: y = (P@V)/l * conf ; location = 1/l
  float* yout = out;                              // [B][C][N]
  float* loc  = out + (size_t)BB * CC * NN;       // [B][1][H][W]
#pragma unroll
  for (int r = 0; r < 8; ++r) {
    const int q = qbase + r + hi * 8;
    const float invl = 1.0f / lrow[r];
    const float conf = mrow[r];
#pragma unroll
    for (int cc = 0; cc < 8; ++cc) {
      const int c = cc * 16 + l;
      yout[((size_t)b * CC + c) * NN + q] = acc[cc][r] * invl * conf;
    }
    if (l == 0) loc[(size_t)b * NN + q] = invl;   // lanes 0 and 16 cover rows r and r+8
  }
}

extern "C" void kernel_launch(void* const* d_in, const int* in_sizes, int n_in,
                              void* d_out, int out_size, void* d_ws, size_t ws_size,
                              hipStream_t stream) {
  const float* m       = (const float*)d_in[0];
  const float* rs      = (const float*)d_in[1];
  const float* theta_w = (const float*)d_in[2];
  const float* theta_b = (const float*)d_in[3];
  const float* phi_w   = (const float*)d_in[4];
  const float* phi_b   = (const float*)d_in[5];
  float* out = (float*)d_out;

  // Workspace layout (~28 MB): p buffers, mean sums, f16 theta/phi/rs
  char* ws = (char*)d_ws;
  size_t off = 0;
  float* pbuf = (float*)(ws + off); off += (size_t)2 * BB * CC * NN * sizeof(float);
  float* means = (float*)(ws + off); off += (size_t)2 * BB * CC * sizeof(float);
  off = (off + 255) & ~(size_t)255;
  _Float16* theta_h = (_Float16*)(ws + off); off += (size_t)BB * NN * CC * sizeof(_Float16);
  _Float16* phi_h   = (_Float16*)(ws + off); off += (size_t)BB * NN * CC * sizeof(_Float16);
  _Float16* rs_h    = (_Float16*)(ws + off); off += (size_t)BB * CC * NN * sizeof(_Float16);

  zero_means_kernel<<<4, 256, 0, stream>>>(means);
  proj_kernel<<<dim3(NN / 128, BB, 2), 256, 0, stream>>>(m, rs, theta_w, theta_b,
                                                         phi_w, phi_b, pbuf, means);
  norm_kernel<<<dim3(NN / 256, BB, 2), 256, 0, stream>>>(rs, pbuf, means,
                                                         theta_h, phi_h, rs_h);
  attn_kernel<<<dim3(NN / 128, BB), 256, 0, stream>>>(theta_h, phi_h, rs_h, out);
}
